// ScaledDotProductAttention_53429393162745
// MI455X (gfx1250) — matrix operand order repaired and checked by reading, not compiled, not run on hardware
//
#include <hip/hip_runtime.h>
#include <math.h>

// ---------------------------------------------------------------------------
// Causal multi-head attention forward for MI455X (gfx1250, wave32, WMMA).
//   B=4, S=2048, D=1024, H=16, d_head=64.  All GEMMs via v_wmma_f32_16x16x32_f16.
// Phase 1: Q/K/V = x @ W (+b) tiled GEMM (fp32 -> f16 conversion on the fly).
//          Q scaled by 1/sqrt(64) at store.  Q,K stored [B][S][D] f16,
//          V stored transposed [B*H][64][S] f16 so PV fragments load contiguously.
// Phase 2: flash-attention, 16 query rows per wave, 32-key tiles, fp32 softmax.
// Workspace: 3 * 8Mi f16 elements = 48 MB in d_ws.
// ---------------------------------------------------------------------------

typedef __attribute__((ext_vector_type(16))) _Float16 v16h;
typedef __attribute__((ext_vector_type(8)))  float    v8f;

#define BB   4
#define SS   2048
#define DD   1024
#define HH   16
#define DH   64
#define MM   (BB * SS)          // 8192
#define QK_SCALE 0.125f         // 1/sqrt(64)
#define LOG2E 1.44269504088896340736f
#define LSTR 40                 // padded LDS row stride in halfs (80B -> conflict-free b128)

union FragU { v16h f; uint4 q[2]; };

static __device__ __forceinline__ v16h ld_frag(const _Float16* p0, const _Float16* p1) {
  FragU u;
  u.q[0] = *(const uint4*)p0;
  u.q[1] = *(const uint4*)p1;
  return u.f;
}

static __device__ __forceinline__ v8f wmma16(v16h a, v16h b, v8f c) {
  return __builtin_amdgcn_wmma_f32_16x16x32_f16(false, a, false, b, (short)0, c,
                                                false, false);
}

// ---------------------------------------------------------------------------
// Phase 1: C[M=8192, N=1024] = x[M,K=1024] @ W[K,N] + bias.  blockIdx.z picks
// which of Q/K/V.  128x128 block tile, 8 waves of 32x64, K-step 32.
// ---------------------------------------------------------------------------
__global__ __launch_bounds__(256)
void qkv_gemm(const float* __restrict__ x,
              const float* __restrict__ Wq, const float* __restrict__ bq,
              const float* __restrict__ Wk, const float* __restrict__ bk,
              const float* __restrict__ Wv, const float* __restrict__ bv,
              _Float16* __restrict__ Qws, _Float16* __restrict__ Kws,
              _Float16* __restrict__ Vtws) {
  const int zz = blockIdx.z;
  const float* __restrict__ W    = (zz == 0) ? Wq : (zz == 1) ? Wk : Wv;
  const float* __restrict__ bias = (zz == 0) ? bq : (zz == 1) ? bk : bv;

  const int m0 = blockIdx.x * 128;
  const int n0 = blockIdx.y * 128;

  __shared__ _Float16 As[128 * LSTR];   // A tile, row-major [m][k], padded
  __shared__ _Float16 Bs[128 * LSTR];   // W tile, TRANSPOSED  [n][k], padded

  const int tid  = threadIdx.x;
  const int wv   = tid >> 5;
  const int lane = tid & 31;
  const int n16  = lane & 15;
  const int half = lane >> 4;
  const int wm0  = (wv >> 1) * 32;      // 4 waves along M
  const int wn0  = (wv & 1) * 64;       // 2 waves along N

  // staging coordinates (computed once)
  const int arow = tid >> 3, aseg = tid & 7;     // + i*32 rows
  const int bkr  = tid >> 5, bnseg = tid & 31;   // + i*8  k-rows

  v8f acc[2][4];
#pragma unroll
  for (int i = 0; i < 2; ++i)
#pragma unroll
    for (int j = 0; j < 4; ++j) acc[i][j] = (v8f){0.f,0.f,0.f,0.f,0.f,0.f,0.f,0.f};

  for (int k0 = 0; k0 < DD; k0 += 32) {
    __syncthreads();
    // ---- prefetch next K-slab into L2/L0 while we stage this one
    if (k0 + 32 < DD) {
      __builtin_prefetch(x + (size_t)(m0 + arow) * DD + (k0 + 32) + aseg * 4, 0, 3);
      __builtin_prefetch(W + (size_t)(k0 + 32 + bkr) * DD + n0 + bnseg * 4, 0, 3);
    }
    // ---- stage A: 128 rows x 32 k  (1024 float4 loads; 4 per thread)
#pragma unroll
    for (int i = 0; i < 4; ++i) {
      int row = arow + i * 32;
      float4 f = *(const float4*)(x + (size_t)(m0 + row) * DD + k0 + aseg * 4);
      union { _Float16 h[4]; uint2 u; } pk;
      pk.h[0] = (_Float16)f.x; pk.h[1] = (_Float16)f.y;
      pk.h[2] = (_Float16)f.z; pk.h[3] = (_Float16)f.w;
      *(uint2*)&As[row * LSTR + aseg * 4] = pk.u;
    }
    // ---- stage B transposed: W rows k0..k0+31, cols n0..n0+127 -> Bs[n][k]
#pragma unroll
    for (int i = 0; i < 4; ++i) {
      int kr = bkr + i * 8;
      float4 f = *(const float4*)(W + (size_t)(k0 + kr) * DD + n0 + bnseg * 4);
      Bs[(bnseg * 4 + 0) * LSTR + kr] = (_Float16)f.x;
      Bs[(bnseg * 4 + 1) * LSTR + kr] = (_Float16)f.y;
      Bs[(bnseg * 4 + 2) * LSTR + kr] = (_Float16)f.z;
      Bs[(bnseg * 4 + 3) * LSTR + kr] = (_Float16)f.w;
    }
    __syncthreads();

    // ---- A fragments (A layout: lane=M, K = 16*half .. 16*half+15 contiguous)
    v16h af[2];
#pragma unroll
    for (int mf = 0; mf < 2; ++mf) {
      const _Float16* pa = &As[(wm0 + mf * 16 + n16) * LSTR + 16 * half];
      af[mf] = ld_frag(pa, pa + 8);
    }
    // ---- B fragments (B layout: lane=N, K = 16*half .. 16*half+15 contiguous)
#pragma unroll
    for (int nf = 0; nf < 4; ++nf) {
      const _Float16* pb = &Bs[(wn0 + nf * 16 + n16) * LSTR + 16 * half];
      v16h bf = ld_frag(pb, pb + 8);
      acc[0][nf] = wmma16(af[0], bf, acc[0][nf]);
      acc[1][nf] = wmma16(af[1], bf, acc[1][nf]);
    }
  }

  // ---- epilogue: +bias, convert f16, store.  zz is block-uniform: branch ONCE.
  const int mbase = m0 + wm0 + 8 * half;          // + mf*16 + v
  const int nbase = n0 + wn0 + n16;               // + nf*16
  float bv4[4];
#pragma unroll
  for (int nf = 0; nf < 4; ++nf) bv4[nf] = bias[nbase + nf * 16];

  if (zz == 0) {
    _Float16* outp = Qws;
#pragma unroll
    for (int nf = 0; nf < 4; ++nf)
#pragma unroll
      for (int mf = 0; mf < 2; ++mf)
#pragma unroll
        for (int v = 0; v < 8; ++v) {
          int mg = mbase + mf * 16 + v;
          outp[(size_t)mg * DD + nbase + nf * 16] =
              (_Float16)((acc[mf][nf][v] + bv4[nf]) * QK_SCALE);
        }
  } else if (zz == 1) {
    _Float16* outp = Kws;
#pragma unroll
    for (int nf = 0; nf < 4; ++nf)
#pragma unroll
      for (int mf = 0; mf < 2; ++mf)
#pragma unroll
        for (int v = 0; v < 8; ++v) {
          int mg = mbase + mf * 16 + v;
          outp[(size_t)mg * DD + nbase + nf * 16] =
              (_Float16)(acc[mf][nf][v] + bv4[nf]);
        }
  } else {
#pragma unroll
    for (int nf = 0; nf < 4; ++nf) {
      int ng = nbase + nf * 16;
      int hh = ng >> 6, dc = ng & 63;
      _Float16* colp = Vtws + ((size_t)hh * DH + dc) * SS;  // + (bb*HH)*DH*SS later
#pragma unroll
      for (int mf = 0; mf < 2; ++mf)
#pragma unroll
        for (int v = 0; v < 8; ++v) {
          int mg = mbase + mf * 16 + v;
          int bb = mg >> 11, s = mg & 2047;
          colp[(size_t)bb * (HH * DH * SS) + s] = (_Float16)(acc[mf][nf][v] + bv4[nf]);
        }
    }
  }
}

// ---------------------------------------------------------------------------
// Phase 2: flash attention.  block = 128 thr (4 waves); wave w handles query
// rows [q0, q0+16) of head (b,h).  grid = (S/64, H, B).
// ---------------------------------------------------------------------------
__global__ __launch_bounds__(128)
void attn_flash(const _Float16* __restrict__ Qws,
                const _Float16* __restrict__ Kws,
                const _Float16* __restrict__ Vtws,
                float* __restrict__ Out) {
  const int b  = blockIdx.z;
  const int h  = blockIdx.y;
  const int qt = blockIdx.x;

  const int tid  = threadIdx.x;
  const int wv   = tid >> 5;
  const int lane = tid & 31;
  const int n16  = lane & 15;
  const int half = lane >> 4;
  const int q0   = qt * 64 + wv * 16;

  // per-wave P staging (C-layout -> A-layout transpose through LDS)
  __shared__ _Float16 Pst[4 * 16 * LSTR];
  _Float16* mine = &Pst[wv * 16 * LSTR];

  // ---- preload Q fragments (16 rows x 64 d = 2 A-frags), already scaled
  v16h qf[2];
#pragma unroll
  for (int dc = 0; dc < 2; ++dc) {
    const _Float16* p =
        Qws + (size_t)(b * SS + q0 + n16) * DD + h * DH + dc * 32 + 16 * half;
    qf[dc] = ld_frag(p, p + 8);
  }

  v8f o[4];
#pragma unroll
  for (int i = 0; i < 4; ++i) o[i] = (v8f){0.f,0.f,0.f,0.f,0.f,0.f,0.f,0.f};
  float mrow[8], lrow[8];
#pragma unroll
  for (int v = 0; v < 8; ++v) { mrow[v] = -INFINITY; lrow[v] = 0.f; }

  const _Float16* Kbase  = Kws + (size_t)b * SS * DD + h * DH;
  const _Float16* Vtbase = Vtws + (size_t)(b * HH + h) * DH * SS;

  for (int j0 = 0; j0 <= q0 + 15; j0 += 32) {
    // ---- scores S[16 x 32] = Q @ K^T  (two 16-col fragments)
    v8f s0 = (v8f){0.f,0.f,0.f,0.f,0.f,0.f,0.f,0.f};
    v8f s1 = s0;
#pragma unroll
    for (int dc = 0; dc < 2; ++dc) {
      // B frag for K^T: lane = key t, contraction over d contiguous in memory
      const _Float16* pk0 =
          Kbase + (size_t)(j0 + n16) * DD + dc * 32 + 16 * half;
      v16h kf0 = ld_frag(pk0, pk0 + 8);
      s0 = wmma16(qf[dc], kf0, s0);
      const _Float16* pk1 =
          Kbase + (size_t)(j0 + 16 + n16) * DD + dc * 32 + 16 * half;
      v16h kf1 = ld_frag(pk1, pk1 + 8);
      s1 = wmma16(qf[dc], kf1, s1);
    }

    // ---- causal mask (C layout: col = n16, row = v + 8*half)
    const int t0 = j0 + n16, t1 = t0 + 16;
#pragma unroll
    for (int v = 0; v < 8; ++v) {
      int qrow = q0 + v + 8 * half;
      if (t0 > qrow) s0[v] = -INFINITY;
      if (t1 > qrow) s1[v] = -INFINITY;
    }

    // ---- online softmax (row stats reduced over the 16-lane N group)
    float alpha[8], p0v[8], p1v[8];
#pragma unroll
    for (int v = 0; v < 8; ++v) {
      float c = fmaxf(s0[v], s1[v]);
      c = fmaxf(c, __shfl_xor(c, 1, 32));
      c = fmaxf(c, __shfl_xor(c, 2, 32));
      c = fmaxf(c, __shfl_xor(c, 4, 32));
      c = fmaxf(c, __shfl_xor(c, 8, 32));
      float mn = fmaxf(mrow[v], c);
      alpha[v] = exp2f((mrow[v] - mn) * LOG2E);
      mrow[v] = mn;
      p0v[v] = exp2f((s0[v] - mn) * LOG2E);
      p1v[v] = exp2f((s1[v] - mn) * LOG2E);
      float rs = p0v[v] + p1v[v];
      rs += __shfl_xor(rs, 1, 32);
      rs += __shfl_xor(rs, 2, 32);
      rs += __shfl_xor(rs, 4, 32);
      rs += __shfl_xor(rs, 8, 32);
      lrow[v] = lrow[v] * alpha[v] + rs;
    }
#pragma unroll
    for (int nf = 0; nf < 4; ++nf)
#pragma unroll
      for (int v = 0; v < 8; ++v) o[nf][v] *= alpha[v];

    // ---- P (C layout) -> LDS -> A layout fragment
#pragma unroll
    for (int v = 0; v < 8; ++v) {
      int row = v + 8 * half;
      mine[row * LSTR + n16]      = (_Float16)p0v[v];
      mine[row * LSTR + 16 + n16] = (_Float16)p1v[v];
    }
    const _Float16* pp = &mine[n16 * LSTR + 16 * half];
    v16h pf = ld_frag(pp, pp + 8);

    // ---- O += P @ V   (V transposed in ws: contraction t contiguous)
#pragma unroll
    for (int nf = 0; nf < 4; ++nf) {
      const _Float16* pv =
          Vtbase + (size_t)(nf * 16 + n16) * SS + j0 + 16 * half;
      v16h vf = ld_frag(pv, pv + 8);
      o[nf] = wmma16(pf, vf, o[nf]);
    }
  }

  // ---- normalize and store fp32
  float inv[8];
#pragma unroll
  for (int v = 0; v < 8; ++v) inv[v] = 1.0f / lrow[v];
#pragma unroll
  for (int nf = 0; nf < 4; ++nf) {
#pragma unroll
    for (int v = 0; v < 8; ++v) {
      int row = q0 + v + 8 * half;
      int col = h * DH + nf * 16 + n16;
      Out[(size_t)(b * SS + row) * DD + col] = o[nf][v] * inv[v];
    }
  }
}

// ---------------------------------------------------------------------------
extern "C" void kernel_launch(void* const* d_in, const int* in_sizes, int n_in,
                              void* d_out, int out_size, void* d_ws, size_t ws_size,
                              hipStream_t stream) {
  const float* x  = (const float*)d_in[0];
  const float* Wq = (const float*)d_in[1];
  const float* bq = (const float*)d_in[2];
  const float* Wk = (const float*)d_in[3];
  const float* bk = (const float*)d_in[4];
  const float* Wv = (const float*)d_in[5];
  const float* bv = (const float*)d_in[6];
  float* out = (float*)d_out;

  const size_t elems = (size_t)MM * DD;            // 8Mi per matrix
  _Float16* Qws  = (_Float16*)d_ws;
  _Float16* Kws  = Qws + elems;
  _Float16* Vtws = Kws + elems;                    // needs 48 MB of ws

  dim3 g1(MM / 128, DD / 128, 3);                  // 64 x 8 x 3
  qkv_gemm<<<g1, 256, 0, stream>>>(x, Wq, bq, Wk, bk, Wv, bv, Qws, Kws, Vtws);

  dim3 g2(SS / 64, HH, BB);                        // 32 x 16 x 4
  attn_flash<<<g2, 128, 0, stream>>>(Qws, Kws, Vtws, out);
}
